// MS1_25829933318797
// MI455X (gfx1250) — matrix operand order repaired and checked by reading, not compiled, not run on hardware
//
#include <hip/hip_runtime.h>
#include <math.h>

typedef __bf16 bf16_t;
typedef __attribute__((ext_vector_type(16))) __bf16 v16bf;
typedef __attribute__((ext_vector_type(8)))  __bf16 v8bf;
typedef __attribute__((ext_vector_type(8)))  float  v8f;
typedef __attribute__((ext_vector_type(4)))  float  v4f;
typedef __attribute__((ext_vector_type(4)))  unsigned int v4u;
typedef __attribute__((ext_vector_type(8)))  int    v8i;
typedef __attribute__((ext_vector_type(4)))  int    v4i;

#define N_LAYERS 64
#define NF       512
#define NF2      256
#define BATCH    8192
#define MTILE    32
#define HSTEP    (1.0f/64.0f)
#define KLAYER   (NF2*NF2)          // 65536 elements per layer

// LDS layout (dynamic shared memory assumed to start at LDS offset 0).
// K buffers use row stride 264 bf16 (256 data + 16B pad -> bank-conflict-free
// b128 fragment loads; matches TDM pad_interval=128dw, pad_amount=4dw).
#define KSTR 264
#define SSTR 264
#define OFF_KA 0u
#define OFF_KT 135168u              // 256*264*2
#define OFF_YB 270336u
#define OFF_ZB 287232u              // + 32*264*2
#define LDS_BYTES 304128u           // 297 KB < 320 KB WGP LDS

#if defined(__has_builtin)
# if __has_builtin(__builtin_amdgcn_tensor_load_to_lds)
#  define HAVE_TDM 1
# endif
#endif
#ifndef HAVE_TDM
# define HAVE_TDM 0
#endif

__device__ __forceinline__ float fast_tanh(float x) {
#if defined(__has_builtin) && __has_builtin(__builtin_amdgcn_tanhf)
  return __builtin_amdgcn_tanhf(x);
#elif defined(__has_builtin) && __has_builtin(__builtin_amdgcn_tanh_f32)
  return __builtin_amdgcn_tanh_f32(x);
#else
  return tanhf(x);
#endif
}

#if HAVE_TDM
// Build a Tensor DMA Descriptor (ISA 08_async_tensor §8.3/8.4) for a 2D
// 256x256 bf16 tile, row stride 256, with LDS padding: 16B pad per 512B row
// (pad_interval code 6 = 128 dwords, pad_amount code 3 = 4 dwords) so the
// LDS image has row stride 264 bf16.
__device__ __forceinline__ void tdm_issue_k(unsigned ldsOff, const bf16_t* gsrc) {
  unsigned long long ga = (unsigned long long)gsrc;
  v4u g0 = {
    1u,                                               // count=1, user mode
    ldsOff,                                           // lds_addr
    (unsigned)(ga & 0xFFFFFFFFull),                   // global_addr[31:0]
    (unsigned)((ga >> 32) & 0x1FFFFFFull) | (2u<<30)  // global_addr[56:32] | type=2
  };
  v8i g1 = {
    (int)((1u<<16) | (1u<<20) | (6u<<22) | (3u<<25)), // data_size=2B, pad_en, int=128dw, amt=4dw
    (int)(((unsigned)NF2 & 0xFFFFu) << 16),           // tensor_dim0 lo16 @ [63:48]
    (int)((((unsigned)NF2 >> 16) & 0xFFFFu) | (((unsigned)NF2 & 0xFFFFu) << 16)), // dim0 hi | dim1 lo
    (int)((((unsigned)NF2 >> 16) & 0xFFFFu) | (((unsigned)NF2 & 0xFFFFu) << 16)), // dim1 hi | tile_dim0
    (int)((unsigned)NF2 & 0xFFFFu),                   // tile_dim1 | tile_dim2=0
    (int)NF2,                                         // tensor_dim0_stride lo32
    0, 0
  };
  v4i gz = {0, 0, 0, 0};
#if __has_include(<hip/amd_detail/amd_gfx1250_TDM.h>)
  v8i gz8 = {0, 0, 0, 0, 0, 0, 0, 0};
  __builtin_amdgcn_tensor_load_to_lds(g0, g1, gz, gz, gz8, 0);
#else
  __builtin_amdgcn_tensor_load_to_lds(g0, g1, gz, gz, 0);
#endif
}
#endif

// Stage one layer of K into LDS. TDM path: wave0 issues an async tensor DMA.
// Fallback: all 256 threads do a synchronous strided copy.
__device__ __forceinline__ void stage_K(bf16_t* dstLds, unsigned ldsOff,
                                        const bf16_t* gsrc, bool leader) {
#if HAVE_TDM
  (void)dstLds;
  if (leader) tdm_issue_k(ldsOff, gsrc);
#else
  (void)ldsOff; (void)leader;
  for (int i = threadIdx.x; i < NF2 * 32; i += 256) {   // 32 x 16B chunks per row
    int r = i >> 5, q = i & 31;
    ((v4u*)(dstLds + r * KSTR))[q] = ((const v4u*)(gsrc + r * NF2))[q];
  }
#endif
}

__device__ __forceinline__ void fence_K(bool leader) {
#if HAVE_TDM
  if (leader) {
# if defined(__has_builtin) && __has_builtin(__builtin_amdgcn_s_wait_tensorcnt)
    __builtin_amdgcn_s_wait_tensorcnt(0);
# else
    asm volatile("s_wait_tensorcnt 0x0" ::: "memory");
# endif
  }
#else
  (void)leader;
#endif
}

// A fragment, 16x32 bf16 (ISA layout: lanes0-15 k in {g0..g0+7, g0+16..g0+23}).
__device__ __forceinline__ v16bf frag_a(const bf16_t* arow, int g0) {
  v8bf lo = *(const v8bf*)(arow + g0);
  v8bf hi = *(const v8bf*)(arow + g0 + 16);
  return __builtin_shufflevector(lo, hi, 0,1,2,3,4,5,6,7,8,9,10,11,12,13,14,15);
}

// B fragment, 32x16 bf16 (lane = n; k in {g0..g0+7, g0+16..g0+23}, g0 = kb + half*8).
__device__ __forceinline__ v16bf frag_b(const bf16_t* kbuf, int nbase, int kb,
                                        int ln, int half) {
  const bf16_t* p = kbuf + (nbase + ln) * KSTR + kb + half * 8;
  v8bf lo = *(const v8bf*)p;
  v8bf hi = *(const v8bf*)(p + 16);
  return __builtin_shufflevector(lo, hi, 0,1,2,3,4,5,6,7,8,9,10,11,12,13,14,15);
}

__global__ __launch_bounds__(256, 1)
void hamnet(const float* __restrict__ Y0,
            const bf16_t* __restrict__ Kbf,   // [l][i][j] = K[i][j][l]
            const bf16_t* __restrict__ KbfT,  // [l][j][i] = K[i][j][l]
            const float* __restrict__ bL,     // [l][512]
            float* __restrict__ out)
{
  extern __shared__ __align__(16) char smem[];
  bf16_t* KA  = (bf16_t*)(smem + OFF_KA);
  bf16_t* KT  = (bf16_t*)(smem + OFF_KT);
  bf16_t* Ybf = (bf16_t*)(smem + OFF_YB);
  bf16_t* Zbf = (bf16_t*)(smem + OFF_ZB);

  const int tid  = threadIdx.x;
  const int lane = tid & 31;
  const int wid  = tid >> 5;
  const int half = lane >> 4;
  const int ln   = lane & 15;
  const int rb   = wid & 1;        // row block (16 rows)
  const int cg   = wid >> 1;       // column group (64 cols)
  const int row0 = blockIdx.x * MTILE;
  const int rbase = rb * 16 + half * 8;
  const int cbase = cg * 64;

  const bool leader = (__builtin_amdgcn_readfirstlane(tid) < 32);

  // f32 master state lives in registers; lane ownership matches the WMMA C/D
  // layout (row = rbase+v, col = cbase+nt*16+ln) and is fixed for all layers.
  float yreg[4][8], zreg[4][8];
  #pragma unroll
  for (int nt = 0; nt < 4; ++nt) {
    const int col = cbase + nt * 16 + ln;
    #pragma unroll
    for (int v = 0; v < 8; ++v) {
      const int r = rbase + v;
      const float* src = Y0 + (size_t)(row0 + r) * NF;
      float y = src[col];
      float z = src[col + NF2];
      yreg[nt][v] = y;
      zreg[nt][v] = z;
      Ybf[r * SSTR + col] = (bf16_t)y;
      Zbf[r * SSTR + col] = (bf16_t)z;
    }
  }

  // Prolog: KA <- K_0 (wait), KT <- K_0^T (left in flight).
  stage_K(KA, OFF_KA, Kbf, leader);
  fence_K(leader);
  stage_K(KT, OFF_KT, KbfT, leader);
  __syncthreads();

  for (int l = 0; l < N_LAYERS; ++l) {
    // ---------------- GEMM1: T1 = Y @ K_l^T  (B fragments from KA) ---------
    {
      v8f acc[4] = {};
      const bf16_t* Arow = Ybf + (rb * 16 + ln) * SSTR;
      #pragma unroll
      for (int kb = 0; kb < NF2; kb += 32) {
        v16bf a = frag_a(Arow, kb + half * 8);
        #pragma unroll
        for (int nt = 0; nt < 4; ++nt) {
          v16bf b = frag_b(KA, cbase + nt * 16, kb, ln, half);
          acc[nt] = __builtin_amdgcn_wmma_f32_16x16x32_bf16(
              false, a, false, b, (short)0, acc[nt], false, false);
        }
      }
      // Z update: Zn = Z + H * tanh(T1 + b_hi)
      const float* bhi = bL + (size_t)l * NF + NF2;
      #pragma unroll
      for (int nt = 0; nt < 4; ++nt) {
        const int col = cbase + nt * 16 + ln;
        const float bv = bhi[col];
        #pragma unroll
        for (int v = 0; v < 8; ++v) {
          float z = zreg[nt][v] + HSTEP * fast_tanh(acc[nt][v] + bv);
          zreg[nt][v] = z;
          Zbf[(rbase + v) * SSTR + col] = (bf16_t)z;
        }
      }
    }
    __syncthreads();                       // KA consumers + Zbf writers done
    fence_K(leader);                       // KT_l now resident
    if (l + 1 < N_LAYERS)
      stage_K(KA, OFF_KA, Kbf + (size_t)(l + 1) * KLAYER, leader);  // overlaps GEMM2
    __syncthreads();

    // ---------------- GEMM2: T2 = Zn @ K_l  (B fragments from KT) ----------
    {
      v8f acc[4] = {};
      const bf16_t* Arow = Zbf + (rb * 16 + ln) * SSTR;
      #pragma unroll
      for (int kb = 0; kb < NF2; kb += 32) {
        v16bf a = frag_a(Arow, kb + half * 8);
        #pragma unroll
        for (int nt = 0; nt < 4; ++nt) {
          v16bf b = frag_b(KT, cbase + nt * 16, kb, ln, half);
          acc[nt] = __builtin_amdgcn_wmma_f32_16x16x32_bf16(
              false, a, false, b, (short)0, acc[nt], false, false);
        }
      }
      // Y update: Yn = Y + H * tanh(-T2 + b_lo)
      const float* blo = bL + (size_t)l * NF;
      #pragma unroll
      for (int nt = 0; nt < 4; ++nt) {
        const int col = cbase + nt * 16 + ln;
        const float bv = blo[col];
        #pragma unroll
        for (int v = 0; v < 8; ++v) {
          float y = yreg[nt][v] + HSTEP * fast_tanh(bv - acc[nt][v]);
          yreg[nt][v] = y;
          Ybf[(rbase + v) * SSTR + col] = (bf16_t)y;
        }
      }
    }
    __syncthreads();                       // KT consumers + Ybf writers done
    fence_K(leader);                       // KA_{l+1} now resident
    if (l + 1 < N_LAYERS)
      stage_K(KT, OFF_KT, KbfT + (size_t)(l + 1) * KLAYER, leader); // overlaps next GEMM1
    __syncthreads();
  }

  // Epilog: registers -> out (f32), [Y | Z].
  #pragma unroll
  for (int nt = 0; nt < 4; ++nt) {
    const int col = cbase + nt * 16 + ln;
    #pragma unroll
    for (int v = 0; v < 8; ++v) {
      const int r = rbase + v;
      float* dst = out + (size_t)(row0 + r) * NF;
      dst[col]       = yreg[nt][v];
      dst[col + NF2] = zreg[nt][v];
    }
  }
}

// ---- prep kernels -----------------------------------------------------------

// K (256,256,64) f32 -> per-layer bf16 copies: Kbf[l][i][j], KbfT[l][j][i].
__global__ void repack_K(const float* __restrict__ K,
                         bf16_t* __restrict__ Kbf,
                         bf16_t* __restrict__ KbfT) {
  int e = blockIdx.x * 256 + threadIdx.x;   // e = i*16384 + j*64 + l (coalesced read)
  int i = e >> 14;
  int j = (e >> 6) & 255;
  int l = e & 63;
  float v = K[e];
  Kbf [(size_t)l * KLAYER + i * NF2 + j] = (bf16_t)v;
  KbfT[(size_t)l * KLAYER + j * NF2 + i] = (bf16_t)v;
}

// b (512,1,64) f32 -> bL[l][512].
__global__ void repack_b(const float* __restrict__ b, float* __restrict__ bL) {
  int e = blockIdx.x * 256 + threadIdx.x;   // e = i*64 + l
  int i = e >> 6, l = e & 63;
  bL[(size_t)l * NF + i] = b[e];
}

extern "C" void kernel_launch(void* const* d_in, const int* in_sizes, int n_in,
                              void* d_out, int out_size, void* d_ws, size_t ws_size,
                              hipStream_t stream) {
  (void)in_sizes; (void)n_in; (void)out_size; (void)ws_size;
  const float* Y0 = (const float*)d_in[0];
  const float* K  = (const float*)d_in[1];
  const float* b  = (const float*)d_in[2];
  float* out = (float*)d_out;

  bf16_t* Kbf  = (bf16_t*)d_ws;                                   // 8 MiB
  bf16_t* KbfT = (bf16_t*)((char*)d_ws + (size_t)8388608);        // 8 MiB
  float*  bL   = (float*) ((char*)d_ws + (size_t)16777216);       // 128 KiB

  repack_K<<<16384, 256, 0, stream>>>(K, Kbf, KbfT);
  repack_b<<<128, 256, 0, stream>>>(b, bL);
  hamnet<<<BATCH / MTILE, 256, LDS_BYTES, stream>>>(Y0, Kbf, KbfT, bL, out);
}
